// SpatialTranscriptFormer_52072183497313
// MI455X (gfx1250) — matrix operand order repaired and checked by reading, not compiled, not run on hardware
//
#include <hip/hip_runtime.h>
#include <hip/hip_bf16.h>
#include <math.h>

// ---------------------------------------------------------------------------
// Types for WMMA
// ---------------------------------------------------------------------------
typedef __attribute__((ext_vector_type(16))) __bf16 v16bf;
typedef __attribute__((ext_vector_type(8)))  float  v8f;
typedef __attribute__((ext_vector_type(8)))  unsigned int v8u;

// Problem constants (reference: B=2, N=4096, D=512, H=8, K=8 -> k_actual=9)
#define BB   2
#define NN   4096
#define DD   512
#define D3   1536
#define KNN  9
#define MM   (BB * NN)   // 8192 rows

__device__ __forceinline__ unsigned short f2bf(float f) {
    unsigned int u = __float_as_uint(f);
    unsigned int r = u + 0x7FFFu + ((u >> 16) & 1u);   // round-to-nearest-even
    return (unsigned short)(r >> 16);
}

// ---------------------------------------------------------------------------
// Kernel 1: k-NN (9 smallest squared distances per query, LDS-resident coords)
// ---------------------------------------------------------------------------
__global__ __launch_bounds__(256) void knn_kernel(
    const float* __restrict__ coords,   // (B, N, 2)
    int* __restrict__ nn)               // (B, N, 9) batch-local indices
{
    __shared__ float2 cs[NN];           // 32 KB (LDS is 320 KB/WGP)
    const int blocksPerBatch = NN / 256;
    const int b     = blockIdx.x / blocksPerBatch;
    const int chunk = blockIdx.x % blocksPerBatch;

    const float2* cb = (const float2*)(coords + (size_t)b * NN * 2);
    for (int i = threadIdx.x; i < NN; i += 256) cs[i] = cb[i];
    __syncthreads();

    const int q = chunk * 256 + threadIdx.x;
    const float qx = cs[q].x, qy = cs[q].y;

    float dbest[KNN];
    int   ibest[KNN];
#pragma unroll
    for (int j = 0; j < KNN; ++j) { dbest[j] = 3.4e38f; ibest[j] = 0; }

    for (int m = 0; m < NN; ++m) {
        const float dx = qx - cs[m].x;
        const float dy = qy - cs[m].y;
        const float d2 = dx * dx + dy * dy;
        if (d2 < dbest[KNN - 1]) {           // strict < keeps earliest index on ties
            dbest[KNN - 1] = d2; ibest[KNN - 1] = m;
#pragma unroll
            for (int j = KNN - 1; j > 0; --j) {
                if (dbest[j] < dbest[j - 1]) {
                    float td = dbest[j]; dbest[j] = dbest[j - 1]; dbest[j - 1] = td;
                    int   ti = ibest[j]; ibest[j] = ibest[j - 1]; ibest[j - 1] = ti;
                }
            }
        }
    }

    int* o = nn + ((size_t)b * NN + q) * KNN;
#pragma unroll
    for (int j = 0; j < KNN; ++j) o[j] = ibest[j];
}

// ---------------------------------------------------------------------------
// Kernel 2: bf16-WMMA GEMM  C(MxN) = A(MxK) @ B(NxK)^T, fp32 accumulate.
// EPI==1 fuses: C = acc + bias[col] + resid[row,col]
// Block tile 128x128, wave tile 32x64, K-step 32, 256 threads (8 waves).
// ---------------------------------------------------------------------------
#define BK 32
#define LDT 34   // padded LDS row stride (ushorts), keeps 4B alignment

template <int EPI>
__global__ __launch_bounds__(256) void gemm_bf16_wmma(
    const float* __restrict__ A,      // M x K
    const float* __restrict__ Bm,     // N x K
    float* __restrict__ C,            // M x N
    int M, int N, int Kd,
    const float* __restrict__ bias,   // N       (EPI==1)
    const float* __restrict__ resid)  // M x N   (EPI==1)
{
    __shared__ unsigned short As[128][LDT];
    __shared__ unsigned short Bs[128][LDT];

    const int tid  = threadIdx.x;
    const int lane = tid & 31;
    const int wave = tid >> 5;
    const int bm = blockIdx.x * 128;
    const int bn = blockIdx.y * 128;

    const int wm = (wave & 3) * 32;   // wave row offset inside block tile
    const int wn = (wave >> 2) * 64;  // wave col offset inside block tile

    const int h  = lane >> 4;         // lane half (0/1)
    const int ml = lane & 15;

    v8f acc[2][4];
#pragma unroll
    for (int i = 0; i < 2; ++i)
#pragma unroll
        for (int j = 0; j < 4; ++j) acc[i][j] = v8f{};

    // A-matrix fragment: ISA 16-bit A 16x32 layout
    auto loadFragA = [&](int rowBase) -> v16bf {
        v8u u;
        const unsigned short* r = As[rowBase + ml];
#pragma unroll
        for (int p = 0; p < 8; ++p) {
            const int kb = ((p < 4) ? 2 * p : 2 * p + 8) + 8 * h;
            u[p] = *(const unsigned int*)(r + kb);
        }
        return __builtin_bit_cast(v16bf, u);
    };
    // B-matrix fragment: lane = output column, same K order as A (ISA 16-bit layout)
    auto loadFragB = [&](int colBase) -> v16bf {
        v8u u;
        const unsigned short* r = Bs[colBase + ml];
#pragma unroll
        for (int p = 0; p < 8; ++p) {
            const int kb = ((p < 4) ? 2 * p : 2 * p + 8) + 8 * h;
            u[p] = *(const unsigned int*)(r + kb);
        }
        return __builtin_bit_cast(v16bf, u);
    };

    for (int k0 = 0; k0 < Kd; k0 += BK) {
        __syncthreads();
        // Stage 128x32 fp32 tiles of A and B into LDS as bf16 (RNE)
        {
            const int row = tid >> 1;            // 0..127
            const int col = (tid & 1) * 16;      // 0 or 16
            const float* sa = A  + (size_t)(bm + row) * Kd + k0 + col;
            const float* sb = Bm + (size_t)(bn + row) * Kd + k0 + col;
#pragma unroll
            for (int e = 0; e < 16; ++e) As[row][col + e] = f2bf(sa[e]);
#pragma unroll
            for (int e = 0; e < 16; ++e) Bs[row][col + e] = f2bf(sb[e]);
        }
        __syncthreads();

        const v16bf a0 = loadFragA(wm);
        const v16bf a1 = loadFragA(wm + 16);
        v16bf bf[4];
#pragma unroll
        for (int j = 0; j < 4; ++j) bf[j] = loadFragB(wn + 16 * j);

#pragma unroll
        for (int j = 0; j < 4; ++j) {
            acc[0][j] = __builtin_amdgcn_wmma_f32_16x16x32_bf16(
                false, a0, false, bf[j], (short)0, acc[0][j], false, false);
            acc[1][j] = __builtin_amdgcn_wmma_f32_16x16x32_bf16(
                false, a1, false, bf[j], (short)0, acc[1][j], false, false);
        }
    }

    // Epilogue: C/D layout -> col = lane%16, row = r + 8*(lane>=16)
#pragma unroll
    for (int i = 0; i < 2; ++i) {
#pragma unroll
        for (int j = 0; j < 4; ++j) {
            const int col = bn + wn + j * 16 + ml;
#pragma unroll
            for (int r = 0; r < 8; ++r) {
                const int row = bm + wm + i * 16 + h * 8 + r;
                const size_t off = (size_t)row * N + col;
                float v = acc[i][j][r];
                if (EPI == 1) v += bias[col] + resid[off];
                C[off] = v;
            }
        }
    }
}

// ---------------------------------------------------------------------------
// Kernel 3: gather-attention over 9 neighbors + exact-erf GELU.
// One wave per query; lane owns dims [16*lane, 16*lane+16). Head = lane/4.
// ---------------------------------------------------------------------------
__global__ __launch_bounds__(256) void attn_kernel(
    const float* __restrict__ qkv,   // (B*N) x 1536  [q | k | v]
    const int*   __restrict__ nn,    // (B*N) x 9, batch-local
    float*       __restrict__ att)   // (B*N) x 512  (post-GELU)
{
    const int wave = threadIdx.x >> 5;
    const int lane = threadIdx.x & 31;
    const int q    = blockIdx.x * 8 + wave;     // 0..8191
    const int b    = q >> 12;                   // / 4096
    const int d0   = lane * 16;

    const float* qrow = qkv + (size_t)q * D3;
    float qv[16];
#pragma unroll
    for (int e = 0; e < 16; ++e) qv[e] = qrow[d0 + e];

    const int* nnq = nn + (size_t)q * KNN;
    int idxs[KNN];
#pragma unroll
    for (int j = 0; j < KNN; ++j) idxs[j] = b * NN + nnq[j];

    // scores: per-head dot(q, k_j) / sqrt(64)
    float sc[KNN];
#pragma unroll
    for (int j = 0; j < KNN; ++j) {
        const float* krow = qkv + (size_t)idxs[j] * D3 + DD;
        float p = 0.f;
#pragma unroll
        for (int e = 0; e < 16; ++e) p += qv[e] * krow[d0 + e];
        p += __shfl_xor(p, 1, 32);   // reduce within the 4 lanes of this head
        p += __shfl_xor(p, 2, 32);
        sc[j] = p * 0.125f;
    }

    // softmax over 9 (replicated across the head's 4 lanes)
    float mx = sc[0];
#pragma unroll
    for (int j = 1; j < KNN; ++j) mx = fmaxf(mx, sc[j]);
    float sum = 0.f;
#pragma unroll
    for (int j = 0; j < KNN; ++j) { sc[j] = __expf(sc[j] - mx); sum += sc[j]; }
    const float inv = 1.f / sum;

    float out[16];
#pragma unroll
    for (int e = 0; e < 16; ++e) out[e] = 0.f;
#pragma unroll
    for (int j = 0; j < KNN; ++j) {
        const float w = sc[j] * inv;
        const float* vrow = qkv + (size_t)idxs[j] * D3 + 2 * DD;
#pragma unroll
        for (int e = 0; e < 16; ++e) out[e] += w * vrow[d0 + e];
    }

    // exact GELU: 0.5*x*(1+erf(x/sqrt(2)))
    float* orow = att + (size_t)q * DD;
#pragma unroll
    for (int e = 0; e < 16; ++e) {
        const float xg = out[e];
        orow[d0 + e] = 0.5f * xg * (1.f + erff(xg * 0.70710678118654752f));
    }
}

// ---------------------------------------------------------------------------
// Launch
// ---------------------------------------------------------------------------
extern "C" void kernel_launch(void* const* d_in, const int* in_sizes, int n_in,
                              void* d_out, int out_size, void* d_ws, size_t ws_size,
                              hipStream_t stream) {
    const float* x      = (const float*)d_in[0];   // (2,4096,512)
    const float* coords = (const float*)d_in[1];   // (2,4096,2)
    const float* w_qkv  = (const float*)d_in[2];   // (1536,512)
    const float* w_proj = (const float*)d_in[3];   // (512,512)
    const float* b_proj = (const float*)d_in[4];   // (512,)
    float* out = (float*)d_out;                    // (2,4096,512)

    // Workspace layout
    const size_t qkv_bytes = (size_t)MM * D3 * sizeof(float);   // 50.3 MB
    const size_t att_bytes = (size_t)MM * DD * sizeof(float);   // 16.8 MB
    float* qkvw = (float*)d_ws;
    float* attw = (float*)((char*)d_ws + qkv_bytes);
    int*   nnw  = (int*)((char*)d_ws + qkv_bytes + att_bytes);

    // 1) kNN indices
    knn_kernel<<<dim3(BB * (NN / 256)), dim3(256), 0, stream>>>(coords, nnw);

    // 2) Full QKV once: (8192x512) @ (1536x512)^T  -> gather replaces 9x recompute
    gemm_bf16_wmma<0><<<dim3(MM / 128, D3 / 128), dim3(256), 0, stream>>>(
        x, w_qkv, qkvw, MM, D3, DD, nullptr, nullptr);

    // 3) Gather-attention over 9 neighbors + GELU
    attn_kernel<<<dim3(MM / 8), dim3(256), 0, stream>>>(qkvw, nnw, attw);

    // 4) Output projection + bias + residual
    gemm_bf16_wmma<1><<<dim3(MM / 128, DD / 128), dim3(256), 0, stream>>>(
        attw, w_proj, out, MM, DD, DD, b_proj, x);
}
